// MRMBlock_12421045420343
// MI455X (gfx1250) — matrix operand order repaired and checked by reading, not compiled, not run on hardware
//
#include <hip/hip_runtime.h>
#include <hip/hip_bf16.h>
#include <cstdint>

typedef __attribute__((ext_vector_type(16))) _Float16     v16h;
typedef __attribute__((ext_vector_type(8)))  float        v8f;
typedef __attribute__((ext_vector_type(8)))  unsigned int v8u;

#define HH 224
#define WW 224
#define NB 8
#define KCHUNK 480   // K-chunk (multiple of 32) held expanded in LDS

#if defined(__HIP_DEVICE_COMPILE__) && \
    __has_builtin(__builtin_amdgcn_global_load_async_to_lds_b32)
#define HAVE_ASYNC_LDS 1
#else
#define HAVE_ASYNC_LDS 0
#endif

// ---------------------------------------------------------------------------
// helpers
// ---------------------------------------------------------------------------
__device__ __forceinline__ unsigned short f2h(float f) {
  _Float16 h = (_Float16)f;
  unsigned short u;
  __builtin_memcpy(&u, &h, 2);
  return u;
}

// async global->LDS dword copy (gfx1250 GLOBAL_LOAD_ASYNC_TO_LDS_B32)
__device__ __forceinline__ void async_g2l_b32(const float* g, float* l) {
#if HAVE_ASYNC_LDS
  __builtin_amdgcn_global_load_async_to_lds_b32(
      (__attribute__((address_space(1))) int*)(uintptr_t)g,
      (__attribute__((address_space(3))) int*)(unsigned int)(uintptr_t)l,
      0, 0);
#else
  *l = *g;
#endif
}

__device__ __forceinline__ void wait_async_lds() {
#if HAVE_ASYNC_LDS
#if __has_builtin(__builtin_amdgcn_s_wait_asynccnt)
  __builtin_amdgcn_s_wait_asynccnt(0);
#else
  asm volatile("s_wait_asynccnt 0" ::: "memory");
#endif
#endif
}

// ---------------------------------------------------------------------------
// Pack OIHW f32 weights into WMMA B-fragment order (f16).
// pk[ nt*(Kpad/32)*512 + ks*512 + lane*16 + j ]
//   co = nt*16 + lane%16 ; k = ks*32 + (lane/16)*16 + j ; k >= Cin*9 -> 0
// ---------------------------------------------------------------------------
__global__ void pack_w_kernel(const float* __restrict__ w, _Float16* __restrict__ pk,
                              int Cout, int Cin, int Kpad) {
  int total = (Cout / 16) * Kpad * 16;
  int idx = blockIdx.x * blockDim.x + threadIdx.x;
  if (idx >= total) return;
  int nt   = idx / (Kpad * 16);
  int rem  = idx % (Kpad * 16);
  int ks   = rem / 512;
  int lrem = rem % 512;
  int lane = lrem / 16;
  int j    = lrem % 16;
  int co = nt * 16 + (lane & 15);
  int k  = ks * 32 + (lane >> 4) * 16 + j;
  int Ktot = Cin * 9;
  float v = 0.0f;
  if (k < Ktot) {
    int ci = k / 9;
    int rc = k - ci * 9;
    v = w[((size_t)co * Cin + ci) * 9 + rc];
  }
  pk[idx] = (_Float16)v;
}

// ---------------------------------------------------------------------------
// Implicit-GEMM 3x3 SAME conv.
// Block = 64 threads (2 waves) -> 16 pixels x NT*16 channels.
//  Phase A: async-stage zero-padded 3x18xCin f32 patch into LDS (ASYNCcnt).
//  Phase B: per 480-K chunk, expand im2col rows into f16-pair dwords
//           A32[k/2][m] so the hot loop needs only 8 imm-offset ds_load_b32.
//  Phase C: K loop; each wave runs NH unrolled v_wmma_f32_16x16x32_f16.
// ---------------------------------------------------------------------------
template <int NT>
__global__ void conv_wmma_kernel(const float* __restrict__ in,
                                 const _Float16* __restrict__ pk,
                                 const float* __restrict__ bias,
                                 float* __restrict__ outP,
                                 float* __restrict__ outAdd, int choff,
                                 int Cin, int Kpad) {
  constexpr int NH = (NT + 1) / 2;
  const int w0 = blockIdx.x * 16;
  const int y  = blockIdx.y;
  const int b  = blockIdx.z;
  const int tid  = threadIdx.x;      // 0..63
  const int lane = tid & 31;
  const int wv   = tid >> 5;
  const int m  = lane & 15;          // pixel row of A / channel col of B,C,D
  const int hi = lane >> 4;

  extern __shared__ char sraw[];
  float* P = (float*)sraw;                               // [Cin][3][18] f32
  const size_t pbytes = (((size_t)Cin * 54 * 4) + 15) & ~(size_t)15;
  unsigned int* A32 = (unsigned int*)(sraw + pbytes);    // [kc/2][16] u32

  // ---- Phase A: stage input patch (rows y-1..y+1, cols w0-1..w0+16) ----
  const int tot = Cin * 54;
  for (int i = tid; i < tot; i += 64) {
    int ci  = i / 54;
    int rem = i % 54;
    int rr  = rem / 18;
    int cc  = rem % 18;
    int yy = y + rr - 1;
    int xx = w0 + cc - 1;
    if (yy >= 0 && yy < HH && xx >= 0 && xx < WW)
      async_g2l_b32(&in[(((size_t)b * Cin + ci) * HH + yy) * WW + xx], &P[i]);
    else
      P[i] = 0.0f;
  }
  wait_async_lds();
  __syncthreads();

  v8f acc[NH] = {};
  const int Ktot      = Cin * 9;
  const int nstepsTot = Kpad / 32;
  const int ntbase = wv ? NH : 0;

  for (int k0 = 0; k0 < Kpad; k0 += KCHUNK) {
    const int kc = (Kpad - k0) < KCHUNK ? (Kpad - k0) : KCHUNK;

    // ---- Phase B: expand chunk into f16-pair dwords A32[t][m] ----
    for (int t = tid; t < kc / 2; t += 64) {
      int kg0 = k0 + 2 * t;
      int kg1 = kg0 + 1;
      int ci0 = kg0 / 9, r0 = kg0 - ci0 * 9, rr0 = r0 / 3, cc0 = r0 - rr0 * 3;
      int ci1 = kg1 / 9, r1 = kg1 - ci1 * 9, rr1 = r1 / 3, cc1 = r1 - rr1 * 3;
      bool ok0 = kg0 < Ktot, ok1 = kg1 < Ktot;
      const float* p0 = P + ci0 * 54 + rr0 * 18 + cc0;
      const float* p1 = P + ci1 * 54 + rr1 * 18 + cc1;
      unsigned int tmp[16];
#pragma unroll
      for (int mm = 0; mm < 16; ++mm) {
        float f0 = ok0 ? p0[mm] : 0.0f;
        float f1 = ok1 ? p1[mm] : 0.0f;
        tmp[mm] = (unsigned int)f2h(f0) | ((unsigned int)f2h(f1) << 16);
      }
      uint4* dst = (uint4*)(A32 + (size_t)t * 16);
#pragma unroll
      for (int q = 0; q < 4; ++q)
        dst[q] = make_uint4(tmp[4 * q], tmp[4 * q + 1], tmp[4 * q + 2], tmp[4 * q + 3]);
    }
    __syncthreads();

    // ---- Phase C: hot K loop ----
    const int steps = kc / 32;
    for (int ks = 0; ks < steps; ++ks) {
      // A fragment: lane(hi,m) -> dwords at t0+{0..7}, t0 = ks*16 + hi*8
      const unsigned int* ap = A32 + ((size_t)(ks * 16 + hi * 8)) * 16 + m;
      union { v8u u; v16h h; } ab;
      ab.u[0] = ap[0];   ab.u[1] = ap[16];  ab.u[2] = ap[32];  ab.u[3] = ap[48];
      ab.u[4] = ap[64];  ab.u[5] = ap[80];  ab.u[6] = ap[96];  ab.u[7] = ap[112];
      v16h afrag = ab.h;

      const int ksg = (k0 >> 5) + ks;  // global K step
      __builtin_prefetch(pk + ((size_t)ntbase * nstepsTot + ksg + 1) * 512 + lane * 16, 0, 0);
#pragma unroll
      for (int ii = 0; ii < NH; ++ii) {
        int nt = ntbase + ii;
        if (nt > NT - 1) nt = NT - 1;  // wave-uniform clamp: EXEC stays all-1s
        v16h bfrag = *(const v16h*)(pk + ((size_t)nt * nstepsTot + ksg) * 512 + lane * 16);
        acc[ii] = __builtin_amdgcn_wmma_f32_16x16x32_f16(
            false, afrag, false, bfrag, (short)0, acc[ii], false, false);
      }
    }
    __syncthreads();  // protect A32 before next chunk overwrites it
  }

  // ---- epilogue: acc[ii][r] holds (M = r + 8*hi, N = m) ----
  const int wbase = w0 + hi * 8;
#pragma unroll
  for (int ii = 0; ii < NH; ++ii) {
    int ntraw = ntbase + ii;
    if (ntraw >= NT) continue;       // drop the clamped duplicate
    int co = ntraw * 16 + m;
    float bv = bias[co];
    if (outP) {
      float* o = &outP[(((size_t)b * (NT * 16) + co) * HH + y) * WW + wbase];
#pragma unroll
      for (int r = 0; r < 8; ++r) o[r] = acc[ii][r] + bv;
    }
    if (outAdd) {
      float* o = &outAdd[(((size_t)b * 144 + choff + co) * HH + y) * WW + wbase];
#pragma unroll
      for (int r = 0; r < 8; ++r) o[r] += acc[ii][r] + bv;
    }
  }
}

// ---------------------------------------------------------------------------
// Fused 5x5 erosion + dilation (quadratic SE on the fly) + combine:
//   out = dilate(h) - erode(h) + r
// ---------------------------------------------------------------------------
__global__ void morpho_combine_kernel(const float* __restrict__ hsrc,
                                      const float* __restrict__ rsrc,
                                      const float* __restrict__ ek1p,
                                      const float* __restrict__ ek2p,
                                      const float* __restrict__ dk1p,
                                      const float* __restrict__ dk2p,
                                      float* __restrict__ out, size_t total) {
  size_t idx = (size_t)blockIdx.x * blockDim.x + threadIdx.x;
  if (idx >= total) return;
  int xw = (int)(idx % WW);
  size_t t = idx / WW;
  int yh = (int)(t % HH);
  size_t bc = t / HH;

  float ek1 = ek1p[0], ek2 = ek2p[0], dk1 = dk1p[0], dk2 = dk2p[0];
  float einv = -1.0f / (4.0f * (ek1 + ek2));
  float dinv = -1.0f / (4.0f * (dk1 + dk2));

  const float* base = hsrc + bc * (size_t)(HH * WW);
  float ero =  __builtin_inff();
  float dil = -__builtin_inff();
#pragma unroll
  for (int i = 0; i < 5; ++i) {
    int yy = yh + i - 2;
    if (yy < 0 || yy >= HH) continue;
    float cy2 = (float)((i - 2) * (i - 2));
#pragma unroll
    for (int j = 0; j < 5; ++j) {
      int xx = xw + j - 2;
      if (xx < 0 || xx >= WW) continue;
      float cx2 = (float)((j - 2) * (j - 2));
      float v = base[yy * WW + xx];
      ero = fminf(ero, v - (ek1 * cx2 + ek2 * cy2) * einv);
      dil = fmaxf(dil, v + (dk1 * cx2 + dk2 * cy2) * dinv);
    }
  }
  out[idx] = dil - ero + rsrc[idx];
}

// ---------------------------------------------------------------------------
// Host driver
// ---------------------------------------------------------------------------
extern "C" void kernel_launch(void* const* d_in, const int* in_sizes, int n_in,
                              void* d_out, int out_size, void* d_ws, size_t ws_size,
                              hipStream_t stream) {
  const float* x   = (const float*)d_in[0];
  const float* w11 = (const float*)d_in[1];  const float* b11 = (const float*)d_in[2];
  const float* w12 = (const float*)d_in[3];  const float* b12 = (const float*)d_in[4];
  const float* w13 = (const float*)d_in[5];  const float* b13 = (const float*)d_in[6];
  const float* w1r = (const float*)d_in[7];  const float* b1r = (const float*)d_in[8];
  const float* w21 = (const float*)d_in[9];  const float* b21 = (const float*)d_in[10];
  const float* w22 = (const float*)d_in[11]; const float* b22 = (const float*)d_in[12];
  const float* w23 = (const float*)d_in[13]; const float* b23 = (const float*)d_in[14];
  const float* w2r = (const float*)d_in[15]; const float* b2r = (const float*)d_in[16];
  const float* erk1 = (const float*)d_in[17];
  const float* erk2 = (const float*)d_in[18];
  const float* dik1 = (const float*)d_in[19];
  const float* dik2 = (const float*)d_in[20];

  char* ws = (char*)d_ws;
  size_t off = 0;
  auto take = [&](size_t bytes) -> void* {
    void* p = ws + off;
    off += (bytes + 255) & ~(size_t)255;
    return p;
  };
  auto kpad_of = [](int Cin) { return ((Cin * 9 + 31) / 32) * 32; };
  auto pk_halves = [&](int Cout, int Cin) -> size_t {
    return (size_t)(Cout / 16) * kpad_of(Cin) * 16;
  };

  _Float16* pk11 = (_Float16*)take(sizeof(_Float16) * pk_halves(48, 32));
  _Float16* pk12 = (_Float16*)take(sizeof(_Float16) * pk_halves(48, 48));
  _Float16* pk13 = (_Float16*)take(sizeof(_Float16) * pk_halves(48, 48));
  _Float16* pk1r = (_Float16*)take(sizeof(_Float16) * pk_halves(144, 32));
  _Float16* pk21 = (_Float16*)take(sizeof(_Float16) * pk_halves(48, 144));
  _Float16* pk22 = (_Float16*)take(sizeof(_Float16) * pk_halves(48, 48));
  _Float16* pk23 = (_Float16*)take(sizeof(_Float16) * pk_halves(48, 48));
  _Float16* pk2r = (_Float16*)take(sizeof(_Float16) * pk_halves(144, 144));

  const size_t elems144 = (size_t)NB * 144 * HH * WW;
  const size_t elems48  = (size_t)NB * 48  * HH * WW;
  float* hbuf = (float*)take(sizeof(float) * elems144);
  float* rbuf = (float*)take(sizeof(float) * elems144);
  float* t1   = (float*)take(sizeof(float) * elems48);
  float* t2   = (float*)take(sizeof(float) * elems48);

  auto pack = [&](const float* w, _Float16* pk, int Cout, int Cin) {
    int total = (int)pk_halves(Cout, Cin);
    pack_w_kernel<<<(total + 255) / 256, 256, 0, stream>>>(w, pk, Cout, Cin, kpad_of(Cin));
  };
  pack(w11, pk11, 48, 32);
  pack(w12, pk12, 48, 48);
  pack(w13, pk13, 48, 48);
  pack(w1r, pk1r, 144, 32);
  pack(w21, pk21, 48, 144);
  pack(w22, pk22, 48, 48);
  pack(w23, pk23, 48, 48);
  pack(w2r, pk2r, 144, 144);

  auto conv = [&](const float* in, const _Float16* pkw, const float* bias,
                  int Cin, int Cout, float* outP, float* outAdd, int choff) {
    int Kpad = kpad_of(Cin);
    dim3 grid(WW / 16, HH, NB);
    size_t pbytes = (((size_t)Cin * 54 * 4) + 15) & ~(size_t)15;
    int kc = Kpad < KCHUNK ? Kpad : KCHUNK;
    size_t shmem = pbytes + (size_t)kc * 32;   // kc/2 rows * 16 u32
    if (Cout == 48)
      conv_wmma_kernel<3><<<grid, 64, shmem, stream>>>(in, pkw, bias, outP, outAdd,
                                                       choff, Cin, Kpad);
    else
      conv_wmma_kernel<9><<<grid, 64, shmem, stream>>>(in, pkw, bias, outP, outAdd,
                                                       choff, Cin, Kpad);
  };

  // rm_block 1: h = concat(x1,x2,x3) + conv_r(x)
  conv(x,  pk1r, b1r, 32, 144, hbuf,    nullptr, 0);
  conv(x,  pk11, b11, 32, 48,  t1,      hbuf,    0);
  conv(t1, pk12, b12, 48, 48,  t2,      hbuf,    48);
  conv(t2, pk13, b13, 48, 48,  nullptr, hbuf,    96);

  // rm_block 2: r = concat(y1,y2,y3) + conv_r(h)
  conv(hbuf, pk2r, b2r, 144, 144, rbuf,    nullptr, 0);
  conv(hbuf, pk21, b21, 144, 48,  t1,      rbuf,    0);
  conv(t1,   pk22, b22, 48,  48,  t2,      rbuf,    48);
  conv(t2,   pk23, b23, 48,  48,  nullptr, rbuf,    96);

  // fused morphology + combine
  size_t total = elems144;
  morpho_combine_kernel<<<(unsigned)((total + 255) / 256), 256, 0, stream>>>(
      hbuf, rbuf, erk1, erk2, dik1, dik2, (float*)d_out, total);
}
